// WindowAttention_68547678044204
// MI455X (gfx1250) — matrix operand order repaired and checked by reading, not compiled, not run on hardware
//
#include <hip/hip_runtime.h>

// ---------------- problem constants ----------------
#define DIM   384
#define TDIM  1152         // 3*DIM
#define HEADS 12
#define HD    32
#define WS7   7
#define LW    49           // 7*7 tokens per window
#define NWIN  2048         // 32 * 8 * 8 windows
// LDS strides (in halfs); chosen so every 8-byte fragment load is aligned
// and row strides avoid systematic bank conflicts.
#define SPX   388          // sXQ / sK stride  (776B rows, %64 words = 2)
#define SPV   384          // sV stride (only element-wise access)
#define SPP   68           // sP stride
#define SPO   36           // sOh stride

typedef __attribute__((ext_vector_type(16))) _Float16 v16h;
typedef __attribute__((ext_vector_type(4)))  _Float16 v4h;
typedef __attribute__((ext_vector_type(8)))  float    v8f;

union V16 { v16h v; v4h q[4]; _Float16 e[16]; };

__device__ __forceinline__ v8f wmma16(v16h a, v16h b, v8f c) {
  // v_wmma_f32_16x16x32_f16: (neg_a, A, neg_b, B, c_mod, C, reuse_a, reuse_b)
  return __builtin_amdgcn_wmma_f32_16x16x32_f16(false, a, false, b, (short)0, c,
                                                false, false);
}

// A fragment (16x32 f16, M x K): lane L(0-15)=row m0+L holds K [kb..kb+15];
// lanes 16-31 hold K [kb+16..kb+31] (matches the B fragment loaders).
__device__ __forceinline__ v16h load_a(const _Float16* base, int stride, int m0,
                                       int kb, int lane) {
  int row = m0 + (lane & 15);
  int kg  = (lane >> 4) * 16;
  const _Float16* p = base + row * stride + kb + kg;
  V16 u;
  u.q[0] = *(const v4h*)(p);
  u.q[1] = *(const v4h*)(p + 4);
  u.q[2] = *(const v4h*)(p + 8);
  u.q[3] = *(const v4h*)(p + 12);
  return u.v;
}

// B fragment (32x16 f16, K x N) from a K-contiguous row-major source
// (B[k][n] = src[n][k], i.e. GEMM against src^T): lane n reads 16 contiguous
// halfs of row n0+n; lanes 0-15 cover K [kb..kb+15], lanes 16-31 K [kb+16..31].
__device__ __forceinline__ v16h load_b_rows(const _Float16* base, int stride,
                                            int n0, int kb, int lane) {
  int row = n0 + (lane & 15);
  int k   = kb + (lane >> 4) * 16;
  const _Float16* p = base + row * stride + k;
  V16 u;
  u.q[0] = *(const v4h*)(p);
  u.q[1] = *(const v4h*)(p + 4);
  u.q[2] = *(const v4h*)(p + 8);
  u.q[3] = *(const v4h*)(p + 12);
  return u.v;
}

// B fragment from a row-major K x N source (column gather, element loads).
__device__ __forceinline__ v16h load_b_cols(const _Float16* base, int stride,
                                            int kb, int c0, int lane) {
  int c  = c0 + (lane & 15);
  int k0 = kb + (lane >> 4) * 16;
  V16 u;
#pragma unroll
  for (int i = 0; i < 16; ++i) u.e[i] = base[(k0 + i) * stride + c];
  return u.v;
}

__device__ __forceinline__ float red_max16(float v) {
#pragma unroll
  for (int off = 8; off; off >>= 1) v = fmaxf(v, __shfl_xor(v, off, 16));
  return v;
}
__device__ __forceinline__ float red_sum16(float v) {
#pragma unroll
  for (int off = 8; off; off >>= 1) v += __shfl_xor(v, off, 16);
  return v;
}

// ---------------- weight fp32 -> f16 pre-pass ----------------
__global__ void convert_w_kernel(const float* __restrict__ wq_f32,
                                 const float* __restrict__ wp_f32,
                                 _Float16* __restrict__ wq,
                                 _Float16* __restrict__ wp, int nq, int np) {
  int i = blockIdx.x * blockDim.x + threadIdx.x;
  if (i < nq) wq[i] = (_Float16)wq_f32[i];
  if (i < np) wp[i] = (_Float16)wp_f32[i];
}

// ---------------- fused window attention: one block per window ----------------
__global__ __launch_bounds__(256, 1)
void winattn_kernel(const float* __restrict__ x,
                    const _Float16* __restrict__ wqkv,   // [1152][384] f16
                    const float* __restrict__ bqkv,      // [1152]
                    const _Float16* __restrict__ wproj,  // [384][384] f16
                    const float* __restrict__ bproj,     // [384]
                    float* __restrict__ out) {
  __shared__ _Float16 sXQ[64 * SPX];   // X (f16), later overwritten by Q*scale
  __shared__ _Float16 sK [64 * SPX];
  __shared__ _Float16 sV [64 * SPV];
  __shared__ _Float16 sP [64 * SPP];   // per-head unnormalized probs
  __shared__ _Float16 sOh[64 * SPO];   // per-head attention output
  __shared__ float    sRed[64 * 4];    // per-row partial max / sum
  __shared__ float    sRow[64];        // per-row max, then 1/sum

  const int tid  = threadIdx.x;
  const int lane = tid & 31;
  const int wv   = tid >> 5;           // wave id, 0..7 (wave32)
  const int ln15 = lane & 15;
  const int kg   = lane >> 4;

  const int wi = blockIdx.x;
  const int b  = wi >> 6;
  const int wy = (wi >> 3) & 7;
  const int wx = wi & 7;
  const int rowbase = b * 3136 + wy * WS7 * 56 + wx * WS7;

  const _Float16 hz = (_Float16)0.f;
  const float scale = 0.17677669529663687f;  // 32^-0.5

  // ---- load X window (49x384 f32 -> f16, zero-pad rows to 64) ----
  for (int e = tid; e < 64 * 96; e += 256) {
    int r  = e / 96;
    int c4 = (e % 96) * 4;
    v4h h4 = {hz, hz, hz, hz};
    if (r < LW) {
      int gr = rowbase + (r / WS7) * 56 + (r % WS7);
      const float4 f = *(const float4*)(x + (size_t)gr * DIM + c4);
      h4[0] = (_Float16)f.x; h4[1] = (_Float16)f.y;
      h4[2] = (_Float16)f.z; h4[3] = (_Float16)f.w;
    }
    *(v4h*)(sXQ + r * SPX + c4) = h4;
  }
  __syncthreads();

  // ---- QKV GEMM: [64x384] @ [384x1152]^T.  K,V tiles first (192 tiles). ----
  for (int t = wv; t < 192; t += 8) {
    int mt = (t & 3) * 16;
    int nn = t >> 2;              // 0..47
    int n0 = DIM + nn * 16;       // N offset within 1152 (K then V)
    __builtin_prefetch(wqkv + (size_t)(n0 + ln15) * DIM, 0, 0);
    v8f acc = {};
#pragma unroll
    for (int ks = 0; ks < 12; ++ks) {
      v16h a  = load_a(sXQ, SPX, mt, ks * 32, lane);
      v16h bf = load_b_rows(wqkv, DIM, n0, ks * 32, lane);
      acc = wmma16(a, bf, acc);
    }
    float bias = bqkv[n0 + ln15];
    _Float16* dst; int st, cb;
    if (nn < 24) { dst = sK; st = SPX; cb = nn * 16; }
    else         { dst = sV; st = SPV; cb = (nn - 24) * 16; }
#pragma unroll
    for (int v = 0; v < 8; ++v) {
      int row = mt + v + 8 * kg;   // C/D layout: VGPR v -> rows v / v+8
      dst[row * st + cb + ln15] = (_Float16)(acc[v] + bias);
    }
  }

  // Q tiles: compute while sXQ still holds X, hold in VGPRs, then overwrite.
  float qh[12][8];
#pragma unroll
  for (int i = 0; i < 12; ++i) {
    int t  = wv + i * 8;          // 0..95
    int mt = (t & 3) * 16;
    int n0 = (t >> 2) * 16;       // 0..368
    v8f acc = {};
#pragma unroll
    for (int ks = 0; ks < 12; ++ks) {
      v16h a  = load_a(sXQ, SPX, mt, ks * 32, lane);
      v16h bf = load_b_rows(wqkv, DIM, n0, ks * 32, lane);
      acc = wmma16(a, bf, acc);
    }
    float bias = bqkv[n0 + ln15];
#pragma unroll
    for (int v = 0; v < 8; ++v) qh[i][v] = (acc[v] + bias) * scale;
  }
  __syncthreads();   // everyone done reading X
#pragma unroll
  for (int i = 0; i < 12; ++i) {
    int t  = wv + i * 8;
    int mt = (t & 3) * 16;
    int n0 = (t >> 2) * 16;
#pragma unroll
    for (int v = 0; v < 8; ++v)
      sXQ[(mt + v + 8 * kg) * SPX + n0 + ln15] = (_Float16)qh[i][v];
  }
  __syncthreads();

  // ---- attention per head, proj accumulated in registers ----
  v8f pacc[12];
  {
    v8f z = {};
#pragma unroll
    for (int i = 0; i < 12; ++i) pacc[i] = z;
  }

  for (int h = 0; h < HEADS; ++h) {
    const int c0 = h * HD;

    // S = (Q*scale) @ K^T : 16 tiles (64x64), 2 per wave, Kdim=32 = 1 WMMA
    float s2[2][8];
#pragma unroll
    for (int tt = 0; tt < 2; ++tt) {
      int t  = wv + tt * 8;       // 0..15
      int mt = (t >> 2) * 16;
      int nt = t & 3;
      v16h a  = load_a(sXQ, SPX, mt, c0, lane);
      v16h bf = load_b_rows(sK, SPX, nt * 16, c0, lane);
      v8f z = {};
      v8f s = wmma16(a, bf, z);
      int  col  = nt * 16 + ln15;
      bool dead = (col >= LW);
#pragma unroll
      for (int v = 0; v < 8; ++v) {
        float sv = dead ? -1e30f : s[v];
        s2[tt][v] = sv;
        float m = red_max16(sv);
        if (ln15 == 0) sRed[(mt + v + 8 * kg) * 4 + nt] = m;
      }
    }
    __syncthreads();
    if (tid < 64)
      sRow[tid] = fmaxf(fmaxf(sRed[tid * 4], sRed[tid * 4 + 1]),
                        fmaxf(sRed[tid * 4 + 2], sRed[tid * 4 + 3]));
    __syncthreads();

    // exp, partial row sums, write unnormalized P (f16)
#pragma unroll
    for (int tt = 0; tt < 2; ++tt) {
      int t  = wv + tt * 8;
      int mt = (t >> 2) * 16;
      int nt = t & 3;
      int col = nt * 16 + ln15;
#pragma unroll
      for (int v = 0; v < 8; ++v) {
        int row = mt + v + 8 * kg;
        float p = __expf(s2[tt][v] - sRow[row]);
        sP[row * SPP + col] = (_Float16)p;
        float su = red_sum16(p);
        if (ln15 == 0) sRed[row * 4 + nt] = su;
      }
    }
    __syncthreads();
    if (tid < 64) {
      float s = sRed[tid * 4] + sRed[tid * 4 + 1] +
                sRed[tid * 4 + 2] + sRed[tid * 4 + 3];
      sRow[tid] = 1.0f / s;   // fold softmax normalization into O
    }
    __syncthreads();

    // O_h = P @ V_h : 64x32, 8 tiles = exactly 1 per wave, Kdim=64 = 2 WMMA
    {
      int mt = (wv >> 1) * 16;
      int nt = (wv & 1) * 16;
      v8f o = {};
#pragma unroll
      for (int ks = 0; ks < 2; ++ks) {
        v16h a  = load_a(sP, SPP, mt, ks * 32, lane);
        v16h bf = load_b_cols(sV, SPV, ks * 32, c0 + nt, lane);
        o = wmma16(a, bf, o);
      }
#pragma unroll
      for (int v = 0; v < 8; ++v) {
        int row = mt + v + 8 * kg;
        sOh[row * SPO + nt + ln15] = (_Float16)(o[v] * sRow[row]);
      }
    }
    __syncthreads();

    // proj partial: pacc += O_h @ Wp[:, c0:c0+32]^T  (96 tiles, 12/wave)
#pragma unroll
    for (int i = 0; i < 12; ++i) {
      int t  = wv + i * 8;
      int mt = (t & 3) * 16;
      int n0 = (t >> 2) * 16;
      v16h a  = load_a(sOh, SPO, mt, 0, lane);
      v16h bf = load_b_rows(wproj, DIM, n0, c0, lane);
      pacc[i] = wmma16(a, bf, pacc[i]);
    }
    __syncthreads();   // sP/sOh/sRed reused by next head
  }

  // ---- epilogue: +bias, window_reverse scatter to (B, H*W, C) f32 ----
#pragma unroll
  for (int i = 0; i < 12; ++i) {
    int t  = wv + i * 8;
    int mt = (t & 3) * 16;
    int n0 = (t >> 2) * 16;
    int col = n0 + ln15;
    float pb = bproj[col];
#pragma unroll
    for (int v = 0; v < 8; ++v) {
      int row = mt + v + 8 * kg;
      if (row < LW) {
        int gr = rowbase + (row / WS7) * 56 + (row % WS7);
        out[(size_t)gr * DIM + col] = pacc[i][v] + pb;
      }
    }
  }
}

extern "C" void kernel_launch(void* const* d_in, const int* in_sizes, int n_in,
                              void* d_out, int out_size, void* d_ws, size_t ws_size,
                              hipStream_t stream) {
  const float* x      = (const float*)d_in[0];
  const float* qkv_w  = (const float*)d_in[1];
  const float* qkv_b  = (const float*)d_in[2];
  const float* proj_w = (const float*)d_in[3];
  const float* proj_b = (const float*)d_in[4];
  // d_in[5]=H, d_in[6]=W (56/56, statically known)

  _Float16* wq = (_Float16*)d_ws;            // 1152*384 f16
  _Float16* wp = wq + (size_t)TDIM * DIM;    // 384*384 f16  (~1.13 MB total ws)
  const int nq = TDIM * DIM, np = DIM * DIM;

  convert_w_kernel<<<(nq + 255) / 256, 256, 0, stream>>>(qkv_w, proj_w, wq, wp,
                                                         nq, np);
  winattn_kernel<<<NWIN, 256, 0, stream>>>(x, wq, qkv_b, wp, proj_b,
                                           (float*)d_out);
}
